// BiGNN_52956946760189
// MI455X (gfx1250) — compile-verified, matrix operand order checked by reading, not run on hardware
//
#include <hip/hip_runtime.h>

typedef __attribute__((ext_vector_type(16))) _Float16 v16h;
typedef __attribute__((ext_vector_type(8)))  float    v8f;
typedef __attribute__((ext_vector_type(4)))  float    v4f;

#define BN_EPS 1e-3f

// ---------------------------------------------------------------------------
// Pack weights into per-lane WMMA B-fragment order (mirroring the documented
// 16-bit A-matrix 16x32 layout), folding the eval-BN channel scale into the
// weight columns. Also emits the folded per-channel bias t = beta - mean*s.
// frag layout: [kc][nt][lane][16 halfs], half i of lane L for n-tile nt:
//   N = nt*16 + (L&15)
//   K = kc*32 + ((i&8)?16:0) + (L>>4)*8 + (i&7)
// ---------------------------------------------------------------------------
__global__ void pack_weights(const float* __restrict__ W,
                             const float* __restrict__ bn,   // [4,64] g,b,m,v
                             _Float16* __restrict__ frag,
                             float* __restrict__ bias,
                             int KC /* = C/32 */) {
  int tid = blockIdx.x * blockDim.x + threadIdx.x;
  if (tid < 64) {
    float g = bn[tid], b = bn[64 + tid], m = bn[128 + tid], v = bn[192 + tid];
    float s = g * rsqrtf(v + BN_EPS);
    bias[tid] = b - m * s;
  }
  int total = KC * 4 * 32 * 16;
  if (tid >= total) return;
  int i    = tid & 15;
  int lane = (tid >> 4) & 31;
  int nt   = (tid >> 9) & 3;
  int kc   = tid >> 11;
  int ncol = nt * 16 + (lane & 15);
  int krow = kc * 32 + ((i & 8) ? 16 : 0) + ((lane >> 4) << 3) + (i & 7);
  float s = bn[ncol] * rsqrtf(bn[192 + ncol] + BN_EPS);
  frag[tid] = (_Float16)(W[krow * 64 + ncol] * s);
}

// Build the per-lane A fragment for a 32-wide K-chunk at float pointer fr:
// halfs 0..7  <- fr[c0 .. c0+7],  halfs 8..15 <- fr[c0+16 .. c0+23],
// with c0 = (lane>>4)*8  (matches ISA 16-bit A 16x32 layout).
__device__ __forceinline__ v16h load_a_frag(const float* __restrict__ fr, int c0) {
  v4f f0 = *(const v4f*)(fr + c0);
  v4f f1 = *(const v4f*)(fr + c0 + 4);
  v4f f2 = *(const v4f*)(fr + c0 + 16);
  v4f f3 = *(const v4f*)(fr + c0 + 20);
  v16h a;
#pragma unroll
  for (int j = 0; j < 4; ++j) {
    a[j]      = (_Float16)f0[j];
    a[4 + j]  = (_Float16)f1[j];
    a[8 + j]  = (_Float16)f2[j];
    a[12 + j] = (_Float16)f3[j];
  }
  return a;
}

// ---------------------------------------------------------------------------
// Grouper: one wave processes VPW voxels with the B (weight) fragments held
// in registers. Per voxel: A = gathered [16 neighbors x C] (neighbors = WMMA
// M), f32 accumulate, bias, max over the 16 neighbor rows (8 accs in-lane +
// one shfl_xor(16)); ReLU folds into the max-reduction init.
// ---------------------------------------------------------------------------
template <int KC, int VPW>
__global__ void grouper_kernel(const float* __restrict__ feat,
                               const int* __restrict__ idx,
                               const _Float16* __restrict__ wfrag,
                               const float* __restrict__ bias,
                               float* __restrict__ out, int N) {
  const int C = KC * 32;
  int gwave = (int)((blockIdx.x * blockDim.x + threadIdx.x) >> 5);
  int lane  = threadIdx.x & 31;
  int n0 = gwave * VPW;
  if (n0 >= N) return;                     // wave-uniform: EXEC stays all-ones
  int m    = lane & 15;                    // neighbor row (WMMA M)
  int half = lane >> 4;

  // Register-resident weights + bias for this wave
  const v16h* wf = (const v16h*)wfrag;
  v16h b[KC * 4];
#pragma unroll
  for (int i = 0; i < KC * 4; ++i) b[i] = wf[i * 32 + lane];
  float tb[4];
#pragma unroll
  for (int nt = 0; nt < 4; ++nt) tb[nt] = bias[nt * 16 + m];

  int n_end = n0 + VPW;
  if (n_end > N) n_end = N;                // wave-uniform bound
  for (int n = n0; n < n_end; ++n) {
    int row = idx[n * 16 + m];
    const float* fr = feat + (long long)row * C;

    v8f acc[4] = {};
#pragma unroll
    for (int kc = 0; kc < KC; ++kc) {
      v16h a = load_a_frag(fr, kc * 32 + half * 8);
#pragma unroll
      for (int nt = 0; nt < 4; ++nt) {
        acc[nt] = __builtin_amdgcn_wmma_f32_16x16x32_f16(
            false, a, false, b[kc * 4 + nt], (short)0, acc[nt], false, false);
      }
    }

#pragma unroll
    for (int nt = 0; nt < 4; ++nt) {
      float mx = 0.0f;                     // ReLU floor == relu(max(...))
#pragma unroll
      for (int r = 0; r < 8; ++r) mx = fmaxf(mx, acc[nt][r] + tb[nt]);
      mx = fmaxf(mx, __shfl_xor(mx, 16, 32));
      if (lane < 16) out[(long long)n * 64 + nt * 16 + m] = mx;
    }
  }
}

// ---------------------------------------------------------------------------
// Final: one wave per 32 voxels (two stacked 16-row M-tiles). Each loaded B
// fragment is reused for two WMMAs, halving weight-fragment vmem traffic.
// lrx = [feat_last | p1 | p2] (K=192 = 6x32): 6 kc x 4 nt x 2 tiles = 48 WMMA.
// ---------------------------------------------------------------------------
__global__ void final_kernel(const float* __restrict__ feat_last,
                             const float* __restrict__ p1,
                             const float* __restrict__ p2,
                             const _Float16* __restrict__ wfrag,
                             const float* __restrict__ bias,
                             float* __restrict__ out, int N) {
  int gwave = (int)((blockIdx.x * blockDim.x + threadIdx.x) >> 5);
  int lane  = threadIdx.x & 31;
  int n0 = gwave * 32;
  if (n0 >= N) return;                     // wave-uniform
  int m    = lane & 15;                    // voxel row within tile (WMMA M)
  int half = lane >> 4;

  int v0 = n0 + m;       if (v0 >= N) v0 = N - 1;   // clamp loads
  int v1 = n0 + 16 + m;  if (v1 >= N) v1 = N - 1;
  const float* s0[3] = { feat_last + (long long)v0 * 64,
                         p1        + (long long)v0 * 64,
                         p2        + (long long)v0 * 64 };
  const float* s1[3] = { feat_last + (long long)v1 * 64,
                         p1        + (long long)v1 * 64,
                         p2        + (long long)v1 * 64 };
  const v16h* wf = (const v16h*)wfrag;

  v8f acc0[4] = {}, acc1[4] = {};
#pragma unroll
  for (int kc = 0; kc < 6; ++kc) {
    int c0 = (kc & 1) * 32 + half * 8;
    v16h a0 = load_a_frag(s0[kc >> 1], c0);
    v16h a1 = load_a_frag(s1[kc >> 1], c0);
#pragma unroll
    for (int nt = 0; nt < 4; ++nt) {
      v16h bb = wf[(kc * 4 + nt) * 32 + lane];
      acc0[nt] = __builtin_amdgcn_wmma_f32_16x16x32_f16(
          false, a0, false, bb, (short)0, acc0[nt], false, false);
      acc1[nt] = __builtin_amdgcn_wmma_f32_16x16x32_f16(
          false, a1, false, bb, (short)0, acc1[nt], false, false);
    }
  }

#pragma unroll
  for (int nt = 0; nt < 4; ++nt) {
    float tb = bias[nt * 16 + m];
#pragma unroll
    for (int r = 0; r < 8; ++r) {
      int row0 = n0 + r + half * 8;        // C/D layout: lanes>=16 hold M=8+r
      int row1 = row0 + 16;
      if (row0 < N) {
        float x = fmaxf(acc0[nt][r] + tb, 0.0f);
        out[(long long)row0 * 64 + nt * 16 + m] = x;
      }
      if (row1 < N) {
        float x = fmaxf(acc1[nt][r] + tb, 0.0f);
        out[(long long)row1 * 64 + nt * 16 + m] = x;
      }
    }
  }
}

extern "C" void kernel_launch(void* const* d_in, const int* in_sizes, int n_in,
                              void* d_out, int out_size, void* d_ws, size_t ws_size,
                              hipStream_t stream) {
  const float* feat_s1   = (const float*)d_in[0];
  const float* feat_s2   = (const float*)d_in[1];
  const float* feat_last = (const float*)d_in[2];
  const float* Wg1       = (const float*)d_in[3];
  const float* bn_g1     = (const float*)d_in[4];
  const float* Wg2       = (const float*)d_in[5];
  const float* bn_g2     = (const float*)d_in[6];
  const float* Wout      = (const float*)d_in[7];
  const float* bn_out    = (const float*)d_in[8];
  const int*   idx_s1    = (const int*)d_in[9];
  const int*   idx_s2    = (const int*)d_in[10];

  int N = in_sizes[2] / 64;                // feat_last is [N,64]

  // Workspace layout: p1 | p2 | packed weight fragments | biases
  float* p1 = (float*)d_ws;
  float* p2 = p1 + (size_t)N * 64;
  _Float16* wf1 = (_Float16*)(p2 + (size_t)N * 64);  // 1*2048 halfs
  _Float16* wf2 = wf1 + 1 * 2048;                    // 2*2048 halfs
  _Float16* wfo = wf2 + 2 * 2048;                    // 6*2048 halfs
  float* b1 = (float*)(wfo + 6 * 2048);
  float* b2 = b1 + 64;
  float* bo = b2 + 64;

  pack_weights<<<(1 * 2048 + 255) / 256, 256, 0, stream>>>(Wg1, bn_g1, wf1, b1, 1);
  pack_weights<<<(2 * 2048 + 255) / 256, 256, 0, stream>>>(Wg2, bn_g2, wf2, b2, 2);
  pack_weights<<<(6 * 2048 + 255) / 256, 256, 0, stream>>>(Wout, bn_out, wfo, bo, 6);

  const int VPW = 8;                       // voxels per wave in groupers
  int gblocks = (N + 8 * VPW - 1) / (8 * VPW);   // 8 waves/block
  grouper_kernel<1, VPW><<<gblocks, 256, 0, stream>>>(feat_s1, idx_s1, wf1, b1, p1, N);
  grouper_kernel<2, VPW><<<gblocks, 256, 0, stream>>>(feat_s2, idx_s2, wf2, b2, p2, N);

  int fblocks = (N + 255) / 256;           // 8 waves/block, 32 voxels/wave
  final_kernel<<<fblocks, 256, 0, stream>>>(feat_last, p1, p2, wfo, bo,
                                            (float*)d_out, N);
}
